// GATN6_80917183857360
// MI455X (gfx1250) — matrix operand order repaired and checked by reading, not compiled, not run on hardware
//
#include <hip/hip_runtime.h>
#include <hip/hip_bf16.h>

typedef __attribute__((ext_vector_type(16))) __bf16 v16bf;
typedef __attribute__((ext_vector_type(8)))  float  v8f;
typedef __attribute__((ext_vector_type(4)))  unsigned int u32x4;
typedef __attribute__((ext_vector_type(8)))  int    i32x8;
typedef __attribute__((ext_vector_type(4)))  int    i32x4;

#define NEG_SLOPE 0.2f

// ---------------------------------------------------------------------------
// WMMA GEMM: H[N,64] = X[N,din] @ W[din,64]; bf16 WMMA operands, f32 accum.
// Block = 128 threads = 4 waves; block owns a 16-row tile, each wave a 16-col
// tile. Both the X tile and full W are brought into LDS by the Tensor Data
// Mover (wave 0 issues 2 tensor_load_to_lds, waits TENSORcnt, then barrier).
// The X-tile descriptor uses TDM LDS padding (+4 dwords/row) so the staged
// layout has a conflict-free stride of DIN+4 floats.
// ---------------------------------------------------------------------------
template <int DIN>
__global__ __launch_bounds__(128) void gat_gemm_wmma(
    const float* __restrict__ X, const float* __restrict__ W,
    float* __restrict__ H, int N)
{
    constexpr int STRIDE    = DIN + 4;            // floats per LDS row (padded)
    constexpr int XS_FLOATS = 16 * STRIDE;
    __shared__ float smem[XS_FLOATS + DIN * 64];
    float* xs = smem;                              // 16 x DIN (stride DIN+4)
    float* wl = smem + XS_FLOATS;                  // DIN x 64 (row-major)

    const int lane = threadIdx.x & 31;
    const int wave = threadIdx.x >> 5;
    const int hi   = (lane >> 4) & 1;
    const int lo   = lane & 15;
    const int rowBase = blockIdx.x * 16;
    const int colBase = wave * 16;

    if (threadIdx.x < 32) {                        // wave 0: issue TDM loads
        unsigned ldsBase = (unsigned)(unsigned long long)smem;   // LDS byte addr
        unsigned long long xga = (unsigned long long)(X + (size_t)rowBase * DIN);
        unsigned long long wga = (unsigned long long)W;
        i32x4 gz4 = {0, 0, 0, 0};
        i32x8 gz8 = {0, 0, 0, 0, 0, 0, 0, 0};

        // ---- X tile: 16 rows x DIN f32, pad 4 dwords after each DIN dwords ----
        u32x4 g0x; i32x8 g1x;
        g0x[0] = 1u;                                             // count = 1
        g0x[1] = ldsBase;                                        // lds_addr
        g0x[2] = (unsigned)xga;                                  // global_addr lo
        g0x[3] = (unsigned)((xga >> 32) & 0x01FFFFFFull) | 0x80000000u; // hi + type=2
        constexpr unsigned PAD_IV = (DIN == 128) ? 6u : 5u;      // 128 / 64 dwords
        g1x[0] = (int)((2u << 16) | (1u << 20) | (PAD_IV << 22) | (3u << 25));
        g1x[1] = (int)((unsigned)(DIN & 0xFFFF) << 16);          // tensor_dim0 lo16
        g1x[2] = (int)(((unsigned)DIN >> 16) | (((unsigned)N & 0xFFFFu) << 16));
        g1x[3] = (int)(((unsigned)N >> 16) | ((unsigned)DIN << 16)); // tile_dim0=DIN
        g1x[4] = 16;                                             // tile_dim1 = 16
        g1x[5] = DIN;                                            // dim0 stride
        g1x[6] = 0; g1x[7] = 0;
        __builtin_amdgcn_tensor_load_to_lds(g0x, g1x, gz4, gz4, gz8, 0);

        // ---- W: DIN rows x 64 cols f32, contiguous, no padding ----
        u32x4 g0w; i32x8 g1w;
        g0w[0] = 1u;
        g0w[1] = ldsBase + XS_FLOATS * 4u;
        g0w[2] = (unsigned)wga;
        g0w[3] = (unsigned)((wga >> 32) & 0x01FFFFFFull) | 0x80000000u;
        g1w[0] = (int)(2u << 16);                                // data_size = 4B
        g1w[1] = (int)(64u << 16);                               // tensor_dim0 = 64
        g1w[2] = (int)((unsigned)(DIN & 0xFFFF) << 16);          // tensor_dim1 = DIN
        g1w[3] = (int)(((unsigned)DIN >> 16) | (64u << 16));     // tile_dim0 = 64
        g1w[4] = DIN;                                            // tile_dim1 = DIN
        g1w[5] = 64;                                             // dim0 stride
        g1w[6] = 0; g1w[7] = 0;
        __builtin_amdgcn_tensor_load_to_lds(g0w, g1w, gz4, gz4, gz8, 0);

        __builtin_amdgcn_s_wait_tensorcnt(0);
    }
    __syncthreads();

    v8f acc = {};
#pragma unroll
    for (int k0 = 0; k0 < DIN; k0 += 32) {
        v16bf a, b;
#pragma unroll
        for (int j = 0; j < 16; ++j) {
            // A fragment (16x32 bf16), ISA 7.12.2 16-bit A layout
            int ka = (j < 8) ? (j + 8 * hi) : (j + 8 + 8 * hi);
            a[j] = (__bf16)xs[lo * STRIDE + k0 + ka];
            // B fragment: same K order as A (table order)
            int kb = k0 + ka;
            b[j] = (__bf16)wl[kb * 64 + colBase + lo];
        }
        acc = __builtin_amdgcn_wmma_f32_16x16x32_bf16(
            false, a, false, b, (short)0, acc, false, false);
    }
    // C/D layout: VGPR r -> M = r + 8*hi, N = lane&15
#pragma unroll
    for (int r = 0; r < 8; ++r)
        H[(size_t)(rowBase + r + 8 * hi) * 64 + colBase + lo] = acc[r];
}

// ---------------------------------------------------------------------------
// Per-node attention logits: ls = h . a_src, ld = h . a_dst  (dout == 64)
// ---------------------------------------------------------------------------
__global__ __launch_bounds__(256) void gat_logits(
    const float* __restrict__ H, const float* __restrict__ as_,
    const float* __restrict__ ad_, float* __restrict__ ls,
    float* __restrict__ ld, int N)
{
    int node = blockIdx.x * 8 + (threadIdx.x >> 5);
    int lane = threadIdx.x & 31;
    if (node >= N) return;
    const float* h = H + (size_t)node * 64;
    float h0 = h[lane], h1 = h[lane + 32];
    float s = h0 * as_[lane] + h1 * as_[lane + 32];
    float d = h0 * ad_[lane] + h1 * ad_[lane + 32];
#pragma unroll
    for (int off = 16; off > 0; off >>= 1) {
        s += __shfl_xor(s, off, 32);
        d += __shfl_xor(d, off, 32);
    }
    if (lane == 0) { ls[node] = s; ld[node] = d; }
}

// ---------------------------------------------------------------------------
// Zero accum/denom, reset segment-max (order-preserving uint encoding)
// ---------------------------------------------------------------------------
__global__ void gat_init(float* __restrict__ accum, float* __restrict__ denom,
                         unsigned* __restrict__ menc, int N)
{
    int i = blockIdx.x * blockDim.x + threadIdx.x;
    if (i < N * 64) accum[i] = 0.0f;
    if (i < N) { denom[i] = 0.0f; menc[i] = 0u; }
}

__device__ __forceinline__ unsigned enc_f32(float f) {
    unsigned u = __float_as_uint(f);
    return (u & 0x80000000u) ? ~u : (u | 0x80000000u);
}
__device__ __forceinline__ float dec_f32(unsigned u) {
    return __uint_as_float((u & 0x80000000u) ? (u ^ 0x80000000u) : ~u);
}
__device__ __forceinline__ float leaky(float v) {
    return v > 0.0f ? v : NEG_SLOPE * v;
}
__device__ __forceinline__ void atomicAddF32(float* p, float v) {
    __hip_atomic_fetch_add(p, v, __ATOMIC_RELAXED, __HIP_MEMORY_SCOPE_AGENT);
}

// ---------------------------------------------------------------------------
// Pass A over edges (+ self loops): segment max of leaky_relu(ls[src]+ld[dst])
// ---------------------------------------------------------------------------
__global__ void gat_edge_max(const int* __restrict__ ei,
                             const float* __restrict__ ls,
                             const float* __restrict__ ld,
                             unsigned* __restrict__ menc, int E, int N)
{
    int e = blockIdx.x * blockDim.x + threadIdx.x;
    if (e >= E + N) return;
    int s, d;
    if (e < E) { s = ei[e]; d = ei[E + e]; } else { s = d = e - E; }
    atomicMax(menc + d, enc_f32(leaky(ls[s] + ld[d])));
}

// ---------------------------------------------------------------------------
// Pass B over edges: ex = exp(e - m[dst]); denom[dst] += ex;
// accum[dst] += ex * h[src]. 16 lanes per edge, float4 gather, f32 atomics.
// ---------------------------------------------------------------------------
__global__ __launch_bounds__(256) void gat_edge_acc(
    const int* __restrict__ ei, const float* __restrict__ ls,
    const float* __restrict__ ld, const unsigned* __restrict__ menc,
    const float* __restrict__ H, float* __restrict__ accum,
    float* __restrict__ denom, int E, int N)
{
    long long t = (long long)blockIdx.x * blockDim.x + threadIdx.x;
    int e = (int)(t >> 4);
    int g = (int)(t & 15);
    if (e >= E + N) return;
    int s, d;
    if (e < E) { s = ei[e]; d = ei[E + e]; } else { s = d = e - E; }
    float v  = leaky(ls[s] + ld[d]);
    float ex = __expf(v - dec_f32(menc[d]));
    if (g == 0) atomicAddF32(denom + d, ex);
    float4 hv = reinterpret_cast<const float4*>(H + (size_t)s * 64)[g];
    float* ap = accum + (size_t)d * 64 + g * 4;
    atomicAddF32(ap + 0, ex * hv.x);
    atomicAddF32(ap + 1, ex * hv.y);
    atomicAddF32(ap + 2, ex * hv.z);
    atomicAddF32(ap + 3, ex * hv.w);
}

// ---------------------------------------------------------------------------
// Layer epilogues
// ---------------------------------------------------------------------------
__global__ void gat_finalize_relu(const float* __restrict__ accum,
                                  const float* __restrict__ denom,
                                  const float* __restrict__ b,
                                  float* __restrict__ out, int N)
{
    int i = blockIdx.x * blockDim.x + threadIdx.x;
    if (i >= N * 64) return;
    float v = accum[i] / denom[i >> 6] + b[i & 63];
    out[i] = v > 0.0f ? v : 0.0f;
}

__global__ __launch_bounds__(256) void gat_finalize_lsm(
    const float* __restrict__ accum, const float* __restrict__ denom,
    const float* __restrict__ b, float* __restrict__ out, int N)
{
    int node = blockIdx.x * 8 + (threadIdx.x >> 5);
    int lane = threadIdx.x & 31;
    if (node >= N) return;
    float dn = denom[node];
    const float* ap = accum + (size_t)node * 64;
    float v0 = ap[lane] / dn + b[lane];
    float v1 = ap[lane + 32] / dn + b[lane + 32];
    float mx = fmaxf(v0, v1);
#pragma unroll
    for (int off = 16; off > 0; off >>= 1) mx = fmaxf(mx, __shfl_xor(mx, off, 32));
    float se = __expf(v0 - mx) + __expf(v1 - mx);
#pragma unroll
    for (int off = 16; off > 0; off >>= 1) se += __shfl_xor(se, off, 32);
    float lse = __logf(se) + mx;
    out[(size_t)node * 64 + lane]      = v0 - lse;
    out[(size_t)node * 64 + lane + 32] = v1 - lse;
}

// ---------------------------------------------------------------------------
// Host: 6-layer pipeline. ws layout (floats):
//   P: N*64 features (reused per layer) | Q: N*64 linear | R: N*64 accum
//   ls: N | ld: N | denom: N | menc: N (as uint)   => ~78.4 MB
// ---------------------------------------------------------------------------
extern "C" void kernel_launch(void* const* d_in, const int* in_sizes, int n_in,
                              void* d_out, int out_size, void* d_ws, size_t ws_size,
                              hipStream_t stream)
{
    const int N = 100000, E = 1600000;
    const float* x  = (const float*)d_in[0];
    const int*   ei = (const int*)d_in[1];
    // d_in[2] = edge_weight: unused by GATConv(edge_dim=None)

    float* P  = (float*)d_ws;
    float* Q  = P + (size_t)N * 64;
    float* R  = Q + (size_t)N * 64;
    float* ls = R + (size_t)N * 64;
    float* ld = ls + N;
    float* dn = ld + N;
    unsigned* menc = (unsigned*)(dn + N);

    const int rowTiles = N / 16;                // 6250, exact
    const int TE = E + N;                       // edges + self loops
    const int totF = N * 64;

    for (int layer = 0; layer < 6; ++layer) {
        const float* W   = (const float*)d_in[3 + 4 * layer];
        const float* as_ = (const float*)d_in[4 + 4 * layer];
        const float* ad_ = (const float*)d_in[5 + 4 * layer];
        const float* b   = (const float*)d_in[6 + 4 * layer];
        const float* xin = (layer == 0) ? x : P;

        if (layer == 0)
            gat_gemm_wmma<128><<<rowTiles, 128, 0, stream>>>(xin, W, Q, N);
        else
            gat_gemm_wmma<64><<<rowTiles, 128, 0, stream>>>(xin, W, Q, N);

        gat_logits<<<(N + 7) / 8, 256, 0, stream>>>(Q, as_, ad_, ls, ld, N);
        gat_init<<<(totF + 255) / 256, 256, 0, stream>>>(R, dn, menc, N);
        gat_edge_max<<<(TE + 255) / 256, 256, 0, stream>>>(ei, ls, ld, menc, E, N);

        long long accThreads = (long long)TE * 16;
        gat_edge_acc<<<(int)((accThreads + 255) / 256), 256, 0, stream>>>(
            ei, ls, ld, menc, Q, R, dn, E, N);

        if (layer < 5)
            gat_finalize_relu<<<(totF + 255) / 256, 256, 0, stream>>>(R, dn, b, P, N);
        else
            gat_finalize_lsm<<<(N + 7) / 8, 256, 0, stream>>>(R, dn, b, (float*)d_out, N);
    }
}
